// _S2M2_SelfAttn_79379585565379
// MI455X (gfx1250) — matrix operand order repaired and checked by reading, not compiled, not run on hardware
//
#include <hip/hip_runtime.h>

#define DIMC   128
#define HEADS  8
#define INNERC 512
#define HDC    64
#define BATCH  4
#define SEQ    1024
#define PEC    32

typedef __attribute__((ext_vector_type(16))) __bf16 bf16x16;
typedef __attribute__((ext_vector_type(8)))  __bf16 bf16x8;
typedef __attribute__((ext_vector_type(8)))  float  f32x8;
typedef __attribute__((ext_vector_type(2)))  float  f32x2;
typedef __attribute__((ext_vector_type(4)))  unsigned int u32x4;
typedef __attribute__((ext_vector_type(8)))  int    i32x8;
typedef __attribute__((ext_vector_type(4)))  int    i32x4;

#if defined(__has_builtin)
#if __has_builtin(__builtin_amdgcn_tensor_load_to_lds) && \
    __has_builtin(__builtin_amdgcn_s_wait_tensorcnt)
#define USE_TDM 1
#endif
#endif
#ifndef USE_TDM
#define USE_TDM 0
#endif

// A-fragment (16x32 bf16, MxK): per-lane row, 16 contiguous K values
// K(kb..kb+15). Caller passes p = &A[row][kb].
__device__ __forceinline__ bf16x16 frag_a(const __bf16* p) {
  const bf16x8* q = (const bf16x8*)p;
  bf16x8 lo = q[0];
  bf16x8 hi = q[1];
  bf16x16 r;
#pragma unroll
  for (int i = 0; i < 8; ++i) { r[i] = lo[i]; r[i + 8] = hi[i]; }
  return r;
}

// B-fragment (32x16 bf16, KxN): per-lane column, 16 contiguous K values.
__device__ __forceinline__ bf16x16 frag_b(const __bf16* p) {
  const bf16x8* q = (const bf16x8*)p;
  bf16x8 lo = q[0];
  bf16x8 hi = q[1];
  bf16x16 r;
#pragma unroll
  for (int i = 0; i < 8; ++i) { r[i] = lo[i]; r[i + 8] = hi[i]; }
  return r;
}

__device__ __forceinline__ f32x8 wmma_bf16(bf16x16 a, bf16x16 b, f32x8 c) {
  return __builtin_amdgcn_wmma_f32_16x16x32_bf16(false, a, false, b, (short)0, c,
                                                 false, false);
}

#if USE_TDM
// Tensor Data Mover: DMA a 2D tile (tile_w x tile_h, 2-byte elements,
// row stride stride_elems) from global into LDS at lds_addr.
// D# bit layout per CDNA5 ISA ch.8 (group0: flags/lds/global/type=2,
// group1: data_size=1(2B), tensor dims == tile dims, dim0 stride).
__device__ __forceinline__ void tdm_load_2d(unsigned lds_addr, const __bf16* gp,
                                            unsigned tile_w, unsigned tile_h,
                                            unsigned stride_elems) {
  unsigned long long ga = (unsigned long long)(uintptr_t)gp;
  u32x4 g0 = { 1u,                                   // count=1 (user descriptor)
               lds_addr,                             // bits[63:32] lds_addr
               (unsigned)ga,                         // global_addr[31:0]
               (unsigned)((ga >> 32) & 0x01ffffffu) | (2u << 30) }; // addr[56:32] | type=2
  i32x8 g1;
  g1[0] = (int)(1u << 16);                 // workgroup_mask=0 | data_size=1 (2B)
  g1[1] = (int)((tile_w & 0xffffu) << 16); // tensor_dim0[15:0]
  g1[2] = (int)((tile_h & 0xffffu) << 16); // tensor_dim0 hi=0 | tensor_dim1[15:0]
  g1[3] = (int)((tile_w & 0xffffu) << 16); // tensor_dim1 hi=0 | tile_dim0
  g1[4] = (int)(tile_h & 0xffffu);         // tile_dim1 | tile_dim2=0
  g1[5] = (int)stride_elems;               // tensor_dim0_stride[31:0]
  g1[6] = 0;
  g1[7] = 0;
  i32x4 z4 = {0, 0, 0, 0};
#if __has_include(<hip/amd_detail/amd_gfx1250_TDM.h>)
  i32x8 z8 = {0, 0, 0, 0, 0, 0, 0, 0};
  __builtin_amdgcn_tensor_load_to_lds(g0, g1, z4, z4, z8, 0);
#else
  __builtin_amdgcn_tensor_load_to_lds(g0, g1, z4, z4, 0);
#endif
}
#endif

// ---------------------------------------------------------------- convert
__global__ void cvt_bf16_kernel(const float* __restrict__ s,
                                __bf16* __restrict__ d, int n) {
  int i = blockIdx.x * blockDim.x + threadIdx.x;
  if (i < n) d[i] = (__bf16)s[i];
}

// ---------------------------------------------------------------- QKV GEMM
// out[m][n] = sum_k x[m][k] * W[n][k]   (m in [0,4096), n in [0,512))
// q,k written head-split [B*H][N][HD]; v written TRANSPOSED [B*H][HD][N]
// (so attention's PV B-fragments are K-contiguous).
__global__ void __launch_bounds__(32)
qkv_kernel(const __bf16* __restrict__ xb,
           const __bf16* __restrict__ wq, const __bf16* __restrict__ wk,
           const __bf16* __restrict__ wv, const float* __restrict__ bv,
           __bf16* __restrict__ q, __bf16* __restrict__ k,
           __bf16* __restrict__ vt) {
  const int l  = threadIdx.x;
  const int lg = l >> 4, ln = l & 15;
  const int m0 = blockIdx.x * 16, n0 = blockIdx.y * 16, which = blockIdx.z;
  const __bf16* W = (which == 0) ? wq : (which == 1) ? wk : wv;

  const int row = m0 + ln;
  const int n   = n0 + ln;
  f32x8 acc = {};
#pragma unroll
  for (int kc = 0; kc < 4; ++kc) {
    bf16x16 a = frag_a(xb + (size_t)row * DIMC + kc * 32 + lg * 16);
    bf16x16 b = frag_b(W  + (size_t)n   * DIMC + kc * 32 + lg * 16);
    acc = wmma_bf16(a, b, acc);
  }
  const float bias = (which == 2) ? bv[n] : 0.0f;
  const int h = n >> 6, c = n & 63;
#pragma unroll
  for (int r = 0; r < 8; ++r) {
    int m = m0 + r + 8 * lg;
    int bb = m >> 10, i = m & (SEQ - 1);
    size_t bh = (size_t)(bb * HEADS + h);
    __bf16 val = (__bf16)(acc[r] + bias);
    if (which == 0)      q[(bh * SEQ + i) * HDC + c] = val;
    else if (which == 1) k[(bh * SEQ + i) * HDC + c] = val;
    else                 vt[(bh * HDC + c) * SEQ + i] = val;  // transposed
  }
}

// ---------------------------------------------------------------- flash attention
// One wave per (b,h, 16-row query tile).  K / V^T tiles are staged into LDS
// by the Tensor Data Mover (double-buffered, s_wait_tensorcnt-synced);
// falls back to direct global loads if the TDM builtin is unavailable.
__global__ void __launch_bounds__(32)
attn_kernel(const __bf16* __restrict__ qg, const __bf16* __restrict__ kg,
            const __bf16* __restrict__ vtg, const float* __restrict__ pe,
            const float* __restrict__ Wpe, const float* __restrict__ bpe,
            __bf16* __restrict__ attn_out) {
#if USE_TDM
  __shared__ __bf16 lds_k[2][32 * HDC];   // 32 keys x 64 ch, row-major
  __shared__ __bf16 lds_vt[2][HDC * 32];  // 64 ch x 32 keys, row-major
#endif
  __shared__ __bf16 lds_p[16 * 32];       // P tile (bf16) for PV WMMA + pe path
  __shared__ float  lds_ps[16 * 32];      // normalized pe_sum for epilogue

  const int l  = threadIdx.x;
  const int lg = l >> 4, ln = l & 15;
  const int bh = blockIdx.x;
  const int b  = bh / HEADS, h = bh % HEADS;
  const int i0 = blockIdx.y * 16;

  const __bf16* qp  = qg  + (((size_t)bh) * SEQ + i0) * HDC;
  const __bf16* kp  = kg  + ((size_t)bh) * SEQ * HDC;
  const __bf16* vtp = vtg + ((size_t)bh) * HDC * SEQ;

  // Q fragments (row = ln, K-chunks 0..31 and 32..63)
  bf16x16 aQ0 = frag_a(qp + (size_t)ln * HDC + lg * 16);
  bf16x16 aQ1 = frag_a(qp + (size_t)ln * HDC + 32 + lg * 16);

  float m_run[8], l_run[8];
  f32x8 o[4] = {};
  f32x8 pa[2] = {};   // pe accumulators: lane -> channels {2*ln, 2*ln+1}
#pragma unroll
  for (int r = 0; r < 8; ++r) { m_run[r] = -__builtin_inff(); l_run[r] = 0.0f; }

  const float scale = 0.125f;  // HD^-0.5 = 1/8
  const int NCH = SEQ / 32;

#if USE_TDM
  auto issue_tiles = [&](int chunk, int buf) {
    tdm_load_2d((unsigned)(uintptr_t)&lds_k[buf][0],
                kp + (size_t)(chunk * 32) * HDC, HDC, 32, HDC);
    tdm_load_2d((unsigned)(uintptr_t)&lds_vt[buf][0],
                vtp + (size_t)(chunk * 32), 32, HDC, SEQ);
  };
  issue_tiles(0, 0);
#endif

  for (int ch = 0; ch < NCH; ++ch) {
    const int j0  = ch * 32;
    const int buf = ch & 1;
#if USE_TDM
    if (ch + 1 < NCH) {
      issue_tiles(ch + 1, buf ^ 1);
      __builtin_amdgcn_s_wait_tensorcnt(2);  // in-order: this chunk has landed
    } else {
      __builtin_amdgcn_s_wait_tensorcnt(0);
    }
    const __bf16* kt  = &lds_k[buf][0];   // [j_local][c], stride HDC
    const __bf16* vtt = &lds_vt[buf][0];  // [c][j_local], stride 32
    const int VSTRIDE = 32;
    const int JOFF = 0;
#else
    const __bf16* kt  = kp + (size_t)j0 * HDC;
    const __bf16* vtt = vtp;
    const int VSTRIDE = SEQ;
    const int JOFF = j0;
    (void)buf;
#endif

    // ---- scores for two 16-key tiles (K dim = 64 -> 2 WMMAs each)
    f32x8 s0 = {}, s1 = {};
    s0 = wmma_bf16(aQ0, frag_b(kt + (size_t)ln * HDC + lg * 16), s0);
    s0 = wmma_bf16(aQ1, frag_b(kt + (size_t)ln * HDC + 32 + lg * 16), s0);
    s1 = wmma_bf16(aQ0, frag_b(kt + (size_t)(16 + ln) * HDC + lg * 16), s1);
    s1 = wmma_bf16(aQ1, frag_b(kt + (size_t)(16 + ln) * HDC + 32 + lg * 16), s1);

    // ---- online softmax over this 32-key chunk
#pragma unroll
    for (int r = 0; r < 8; ++r) {
      float v0 = s0[r] * scale, v1 = s1[r] * scale;
      float mx = fmaxf(v0, v1);
#pragma unroll
      for (int off = 8; off > 0; off >>= 1)
        mx = fmaxf(mx, __shfl_xor(mx, off, 16));
      float mn = fmaxf(m_run[r], mx);
      float alpha = __expf(m_run[r] - mn);
      float e0 = __expf(v0 - mn), e1 = __expf(v1 - mn);
      float sum = e0 + e1;
#pragma unroll
      for (int off = 8; off > 0; off >>= 1)
        sum += __shfl_xor(sum, off, 16);
      l_run[r] = l_run[r] * alpha + sum;
      m_run[r] = mn;
#pragma unroll
      for (int ct = 0; ct < 4; ++ct) o[ct][r] *= alpha;
      pa[0][r] *= alpha; pa[1][r] *= alpha;
      int rowr = r + 8 * lg;
      lds_p[rowr * 32 + ln]      = (__bf16)e0;
      lds_p[rowr * 32 + 16 + ln] = (__bf16)e1;
    }
    __syncthreads();

    // ---- P·V : A = P (16x32) from LDS, B = V^T rows (K-contiguous)
    bf16x16 aP = frag_a(lds_p + ln * 32 + lg * 16);
#pragma unroll
    for (int ct = 0; ct < 4; ++ct) {
      const __bf16* vb = vtt + (size_t)(ct * 16 + ln) * VSTRIDE + JOFF + lg * 16;
      o[ct] = wmma_bf16(aP, frag_b(vb), o[ct]);
    }

    // ---- pe_sum accumulation: one float2 per (row, key) covers both channels
#pragma unroll
    for (int r = 0; r < 8; ++r) {
      int rowr = r + 8 * lg;
      size_t peb = ((size_t)(i0 + rowr) * SEQ + j0) * PEC + 2 * ln;
      __builtin_prefetch(pe + peb + (size_t)32 * PEC, 0, 1);
      float acc0 = 0.0f, acc1 = 0.0f;
#pragma unroll 4
      for (int jj = 0; jj < 32; ++jj) {
        float p = (float)lds_p[rowr * 32 + jj];
        f32x2 pv = *(const f32x2*)(pe + peb + (size_t)jj * PEC);
        acc0 += p * pv[0];
        acc1 += p * pv[1];
      }
      pa[0][r] += acc0; pa[1][r] += acc1;
    }
    __syncthreads();
  }

  // ---- normalize by row sums, stash pe_sum in LDS for the epilogue
#pragma unroll
  for (int r = 0; r < 8; ++r) {
    float inv = 1.0f / l_run[r];
#pragma unroll
    for (int ct = 0; ct < 4; ++ct) o[ct][r] *= inv;
    pa[0][r] *= inv; pa[1][r] *= inv;
    int rowr = r + 8 * lg;
    lds_ps[rowr * 32 + 2 * ln]     = pa[0][r];
    lds_ps[rowr * 32 + 2 * ln + 1] = pa[1][r];
  }
  __syncthreads();

  // ---- epilogue: out += pe_sum @ Wpe.T + bpe ; store bf16 [B,N,INNER]
#pragma unroll
  for (int ct = 0; ct < 4; ++ct) {
    int c = ct * 16 + ln;
    float bp = bpe[c];
#pragma unroll
    for (int r = 0; r < 8; ++r) {
      int rowr = r + 8 * lg;
      float acc = o[ct][r] + bp;
#pragma unroll 8
      for (int p2 = 0; p2 < PEC; ++p2)
        acc += lds_ps[rowr * 32 + p2] * Wpe[c * PEC + p2];
      attn_out[((size_t)b * SEQ + (i0 + rowr)) * INNERC + h * HDC + c] = (__bf16)acc;
    }
  }
}

// ---------------------------------------------------------------- output proj
// y[m][n] = sum_k attn[m][k] * Wproj[n][k]  (K = 512 -> 16 WMMAs)
__global__ void __launch_bounds__(32)
proj_kernel(const __bf16* __restrict__ ab, const __bf16* __restrict__ wp,
            float* __restrict__ out) {
  const int l  = threadIdx.x;
  const int lg = l >> 4, ln = l & 15;
  const int m0 = blockIdx.x * 16, n0 = blockIdx.y * 16;
  const int row = m0 + ln, n = n0 + ln;
  f32x8 acc = {};
#pragma unroll
  for (int kc = 0; kc < 16; ++kc) {
    bf16x16 a = frag_a(ab + (size_t)row * INNERC + kc * 32 + lg * 16);
    bf16x16 b = frag_b(wp + (size_t)n   * INNERC + kc * 32 + lg * 16);
    acc = wmma_bf16(a, b, acc);
  }
#pragma unroll
  for (int r = 0; r < 8; ++r)
    out[(size_t)(m0 + r + 8 * lg) * DIMC + n0 + ln] = acc[r];
}

// ---------------------------------------------------------------- launch
extern "C" void kernel_launch(void* const* d_in, const int* in_sizes, int n_in,
                              void* d_out, int out_size, void* d_ws, size_t ws_size,
                              hipStream_t stream) {
  const float* x     = (const float*)d_in[0];
  const float* pe    = (const float*)d_in[1];
  const float* Wq    = (const float*)d_in[2];
  const float* Wk    = (const float*)d_in[3];
  const float* Wv    = (const float*)d_in[4];
  const float* bv    = (const float*)d_in[5];
  const float* Wproj = (const float*)d_in[6];
  const float* Wpe   = (const float*)d_in[7];
  const float* bpe   = (const float*)d_in[8];
  float* out = (float*)d_out;

  char* ws = (char*)d_ws;
  size_t off = 0;
  auto take = [&](size_t elems) -> __bf16* {
    __bf16* p = (__bf16*)(ws + off);
    off += ((elems * 2 + 255) / 256) * 256;
    return p;
  };
  __bf16* xb  = take((size_t)BATCH * SEQ * DIMC);   // 4096x128
  __bf16* wqb = take((size_t)INNERC * DIMC);
  __bf16* wkb = take((size_t)INNERC * DIMC);
  __bf16* wvb = take((size_t)INNERC * DIMC);
  __bf16* wpb = take((size_t)DIMC * INNERC);
  __bf16* qb  = take((size_t)BATCH * HEADS * SEQ * HDC);
  __bf16* kb  = take((size_t)BATCH * HEADS * SEQ * HDC);
  __bf16* vtb = take((size_t)BATCH * HEADS * HDC * SEQ);  // transposed V
  __bf16* atb = take((size_t)BATCH * SEQ * INNERC);       // 4096x512
  (void)ws_size; (void)in_sizes; (void)n_in; (void)out_size;

  const int nx = BATCH * SEQ * DIMC;      // 524288
  const int nw = INNERC * DIMC;           // 65536
  cvt_bf16_kernel<<<(nx + 255) / 256, 256, 0, stream>>>(x, xb, nx);
  cvt_bf16_kernel<<<(nw + 255) / 256, 256, 0, stream>>>(Wq, wqb, nw);
  cvt_bf16_kernel<<<(nw + 255) / 256, 256, 0, stream>>>(Wk, wkb, nw);
  cvt_bf16_kernel<<<(nw + 255) / 256, 256, 0, stream>>>(Wv, wvb, nw);
  cvt_bf16_kernel<<<(nw + 255) / 256, 256, 0, stream>>>(Wproj, wpb, nw);

  qkv_kernel<<<dim3((BATCH * SEQ) / 16, INNERC / 16, 3), 32, 0, stream>>>(
      xb, wqb, wkb, wvb, bv, qb, kb, vtb);

  attn_kernel<<<dim3(BATCH * HEADS, SEQ / 16), 32, 0, stream>>>(
      qb, kb, vtb, pe, Wpe, bpe, atb);

  proj_kernel<<<dim3((BATCH * SEQ) / 16, DIMC / 16), 32, 0, stream>>>(
      atb, wpb, out);
}
